// SeqModel_48498770706645
// MI455X (gfx1250) — compile-verified, matrix operand order checked by reading, not run on hardware
//
#include <hip/hip_runtime.h>
#include <hip/hip_fp16.h>

#define HIDDEN 128
#define SEQ    784
#define BATCH  1024
#define NOUT   10
#define MTILE  16
#define TCHUNK 112   // 784 = 7 * 112

typedef __attribute__((ext_vector_type(16))) _Float16 v16h;
typedef __attribute__((ext_vector_type(8)))  float    v8f;

#define NEG_LOG2E (-1.4426950408889634f)

// Map v16h element j -> K offset inside a 32-wide K block, per the CDNA5
// 16-bit A/B fragment layout (lane<16: K = {0..7,16..23}; lane>=16: +8).
__device__ __forceinline__ int frag_k(int j, int half) {
    int v = j >> 1, bit = j & 1;
    return (v < 4 ? 2 * v : 16 + 2 * (v - 4)) + 8 * half + bit;
}

__device__ __forceinline__ float fast_exp2(float x) {
#if __has_builtin(__builtin_amdgcn_exp2f)
    return __builtin_amdgcn_exp2f(x);       // v_exp_f32
#else
    return exp2f(x);
#endif
}

// CDNA5 has a TANH transcendental op; use it if the toolchain exposes it.
__device__ __forceinline__ float fast_tanh(float x) {
#if __has_builtin(__builtin_amdgcn_tanhf)
    return __builtin_amdgcn_tanhf(x);       // v_tanh_f32
#else
    // tanh(x) = 1 - 2 / (exp2(2*log2e*x) + 1), with fast rcp
    return 1.0f - 2.0f * __builtin_amdgcn_rcpf(
        fast_exp2(2.8853900817779268f * x) + 1.0f);
#endif
}

__global__ __launch_bounds__(256, 1)
void gru_scan_kernel(const float* __restrict__ x,
                     const float* __restrict__ Wi,
                     const float* __restrict__ bi,
                     const float* __restrict__ Wh,
                     const float* __restrict__ bh,
                     const float* __restrict__ Wfc,
                     const float* __restrict__ bfc,
                     float* __restrict__ out)
{
    // Ping-pong buffers: one barrier per timestep instead of two.
    __shared__ _Float16 hLds[2][MTILE * HIDDEN];   //  8 KB  f16 shadow of h
    __shared__ float    xLds[2][MTILE * TCHUNK];   // 14 KB  staged x chunks
    __shared__ float    hF32[MTILE * HIDDEN];      //  8 KB  final h for FC head

    const int tid  = threadIdx.x;
    const int wave = tid >> 5;          // 0..7  -> 16-col chunk of hidden dim
    const int lane = tid & 31;
    const int half = lane >> 4;         // 0/1
    const int ln   = lane & 15;         // N (col in tile) / M (row for A)
    const int n    = wave * 16 + ln;    // hidden column 0..127 owned by lane
    const int bb   = blockIdx.x * MTILE;

    // ---- per-lane gate constants (column n of each gate) ----
    // r/z gates are pre-scaled by -log2(e): after the (scaled) WMMA the
    // accumulator is directly the v_exp_f32 argument, no post-mul/add.
    const float swiR = NEG_LOG2E * Wi[n];
    const float swiZ = NEG_LOG2E * Wi[HIDDEN + n];
    const float wiN  = Wi[2 * HIDDEN + n];
    const float scbR = NEG_LOG2E * (bi[n]          + bh[n]);
    const float scbZ = NEG_LOG2E * (bi[HIDDEN + n] + bh[HIDDEN + n]);
    const float biN  = bi[2 * HIDDEN + n];
    const float bhN  = bh[2 * HIDDEN + n];

    // ---- preload Wh^T as f16 B fragments, resident in VGPRs for all 784 steps ----
    // B[k][n] = gscale[g] * Wh[g*128 + n][k]; 3 gates x 4 K-blocks of 32.
    v16h B[3][4];
    #pragma unroll
    for (int g = 0; g < 3; ++g) {
        const float gs = (g < 2) ? NEG_LOG2E : 1.0f;
        const float* whRow = Wh + (size_t)(g * HIDDEN + n) * HIDDEN;
        #pragma unroll
        for (int kb = 0; kb < 4; ++kb) {
            v16h f;
            #pragma unroll
            for (int j = 0; j < 16; ++j)
                f[j] = (_Float16)(gs * whRow[kb * 32 + frag_k(j, half)]);
            B[g][kb] = f;
        }
    }

    // ---- h0 = 0 ----
    float hReg[8];
    #pragma unroll
    for (int v = 0; v < 8; ++v) hReg[v] = 0.0f;
    for (int i = tid; i < MTILE * HIDDEN; i += 256) hLds[0][i] = (_Float16)0.0f;

    int pb = 0;   // h read-buffer parity
    int xb = 0;   // x chunk-buffer parity

    for (int t = 0; t < SEQ; ++t) {
        const int tt = t % TCHUNK;
        if (tt == 0) {
            xb = (t / TCHUNK) & 1;
            // coalesced stage of x[bb:bb+16, t:t+112] into the idle buffer
            for (int i = tid; i < MTILE * TCHUNK; i += 256) {
                int m = i / TCHUNK, c = i % TCHUNK;
                xLds[xb][i] = x[(size_t)(bb + m) * SEQ + t + c];
            }
        }
        __syncthreads();   // single barrier: prev-step h writes + x chunk visible

        // per-lane x for the 8 rows this lane owns; n-gate input term
        // inn = x*wiN + biN computed early (overlaps the A-fragment loads)
        float xv[8], innv[8];
        #pragma unroll
        for (int v = 0; v < 8; ++v) {
            xv[v]   = xLds[xb][(v + 8 * half) * TCHUNK + tt];
            innv[v] = fmaf(xv[v], wiN, biN);
        }

        // A fragments of h from LDS (row-major f16 -> A layout; each 8-element
        // K-run is 16 contiguous bytes -> ds_load_b128)
        v16h A[4];
        #pragma unroll
        for (int kb = 0; kb < 4; ++kb) {
            v16h f;
            #pragma unroll
            for (int j = 0; j < 16; ++j)
                f[j] = hLds[pb][ln * HIDDEN + kb * 32 + frag_k(j, half)];
            A[kb] = f;
        }

        // C init: full pre-activation folded in (scaled) for r/z; bh only for n
        v8f cR, cZ, cN;
        #pragma unroll
        for (int v = 0; v < 8; ++v) {
            cR[v] = fmaf(xv[v], swiR, scbR);
            cZ[v] = fmaf(xv[v], swiZ, scbZ);
            cN[v] = bhN;
        }

        #pragma unroll
        for (int kb = 0; kb < 4; ++kb) {
            cR = __builtin_amdgcn_wmma_f32_16x16x32_f16(false, A[kb], false, B[0][kb], (short)0, cR, false, false);
            cZ = __builtin_amdgcn_wmma_f32_16x16x32_f16(false, A[kb], false, B[1][kb], (short)0, cZ, false, false);
            cN = __builtin_amdgcn_wmma_f32_16x16x32_f16(false, A[kb], false, B[2][kb], (short)0, cN, false, false);
        }

        // gates + state update; write f16 shadow into the *other* h buffer
        // (no WAR hazard -> no second barrier needed)
        #pragma unroll
        for (int v = 0; v < 8; ++v) {
            const int m = v + 8 * half;
            const float r  = __builtin_amdgcn_rcpf(1.0f + fast_exp2(cR[v]));
            const float z  = __builtin_amdgcn_rcpf(1.0f + fast_exp2(cZ[v]));
            const float nn = fast_tanh(fmaf(r, cN[v], innv[v]));
            const float hn = fmaf(z, hReg[v] - nn, nn);   // (1-z)*n + z*h
            hReg[v] = hn;
            hLds[pb ^ 1][m * HIDDEN + n] = (_Float16)hn;
        }
        pb ^= 1;
    }

    // ---- FC head: logits = h @ Wfc^T + bfc (16x10 per workgroup) ----
    __syncthreads();
    #pragma unroll
    for (int v = 0; v < 8; ++v)
        hF32[(v + 8 * half) * HIDDEN + n] = hReg[v];
    __syncthreads();

    for (int i = tid; i < MTILE * NOUT; i += 256) {
        const int m = i / NOUT, o = i % NOUT;
        float acc = bfc[o];
        #pragma unroll 8
        for (int k = 0; k < HIDDEN; ++k)
            acc += hF32[m * HIDDEN + k] * Wfc[o * HIDDEN + k];
        out[(size_t)(bb + m) * NOUT + o] = acc;
    }
}

extern "C" void kernel_launch(void* const* d_in, const int* in_sizes, int n_in,
                              void* d_out, int out_size, void* d_ws, size_t ws_size,
                              hipStream_t stream) {
    (void)in_sizes; (void)n_in; (void)out_size; (void)d_ws; (void)ws_size;
    const float* x   = (const float*)d_in[0];
    const float* Wi  = (const float*)d_in[1];
    const float* bi  = (const float*)d_in[2];
    const float* Wh  = (const float*)d_in[3];
    const float* bh  = (const float*)d_in[4];
    const float* Wfc = (const float*)d_in[5];
    const float* bfc = (const float*)d_in[6];
    float* out = (float*)d_out;

    dim3 grid(BATCH / MTILE);   // 64 workgroups, one 16-row batch tile each
    dim3 block(256);            // 8 wave32 waves -> 8 x 16-col hidden chunks
    gru_scan_kernel<<<grid, block, 0, stream>>>(x, Wi, bi, Wh, bh, Wfc, bfc, out);
}
